// Attention_66082366816924
// MI455X (gfx1250) — matrix operand order repaired and checked by reading, not compiled, not run on hardware
//
#include <hip/hip_runtime.h>

// ---------------------------------------------------------------------------
// MHA forward for MI455X (gfx1250).
//   pass 0: fp32 -> bf16 conversion of x / w_qkv / w_proj (one memory sweep)
//   pass 1: QKV GEMM  (bf16 WMMA, TDM double-buffered LDS tiles)
//   pass 2: flash attention (WMMA + wave32 shuffle softmax)
//   pass 3: proj GEMM + bias (bf16 WMMA, TDM tiles, fp32 out)
//
// Fragment layouts (CDNA5 ISA 7.12.2, wave32):
//   A (16x32, 16-bit): lane l -> row r=l&15; hh=l>>4:
//       us[0..7] = K = hh*8..+7 ; us[8..15] = K = 16+hh*8..+7
//   B (32x16, 16-bit): lane l -> col n=l&15; us[0..7] = K = hh*8..+7 ; us[8..15] = K = 16+hh*8..+7
//   C/D (16x16 f32):   lane l -> col n=l&15; vgpr i -> row m = hh*8 + i
// ---------------------------------------------------------------------------

typedef __attribute__((ext_vector_type(16))) __bf16        bf16x16;
typedef __attribute__((ext_vector_type(8)))  float         f32x8;
typedef __attribute__((ext_vector_type(4)))  float         f32x4;
typedef __attribute__((ext_vector_type(4)))  unsigned int  u32x4;
typedef __attribute__((ext_vector_type(2)))  unsigned int  u32x2;
typedef __attribute__((ext_vector_type(8)))  int           i32x8;
typedef __attribute__((ext_vector_type(4)))  int           i32x4;

union Frag {
  bf16x16 v;
  u32x4   q[2];
};

#define BB   8
#define NN   1024
#define CC   768
#define NH   12
#define HD   64
#define MM   (BB * NN)                         // 8192
#define QKV_ELEMS ((size_t)BB * NH * NN * HD)  // 6291456
#define ATT_ELEMS ((size_t)BB * NN * CC)       // 6291456

#if defined(__has_builtin)
# if __has_builtin(__builtin_amdgcn_tensor_load_to_lds) && \
     __has_builtin(__builtin_amdgcn_s_wait_tensorcnt)
#  define USE_TDM 1
# endif
#endif
#ifndef USE_TDM
# define USE_TDM 0
#endif

__device__ __forceinline__ unsigned short f2bf(float f) {
  unsigned int u = __float_as_uint(f);
  u += 0x7FFFu + ((u >> 16) & 1u);             // round-to-nearest-even
  return (unsigned short)(u >> 16);
}

__device__ __forceinline__ f32x8 wmma_bf16(const Frag& a, const Frag& b, f32x8 c) {
  return __builtin_amdgcn_wmma_f32_16x16x32_bf16(
      false, a.v, false, b.v, (short)0, c, false, false);
}

// Low 32 bits of a flat shared-memory pointer = LDS byte offset (ISA 10.2).
__device__ __forceinline__ unsigned lds_off(const void* p) {
  return (unsigned)(unsigned long long)(__SIZE_TYPE__)p;
}

#if USE_TDM
// 2-D TDM descriptor (ISA 8.3/8.4): bf16 tile tile_rows x 32 from a row-major
// [*, CC] matrix into contiguous LDS.  D# group0: count=1, lds_addr, 57-bit
// global addr, type=2.  group1: data_size=2B, dims/strides/tile dims.
__device__ __forceinline__ void tdm_load_tile(const unsigned short* gp,
                                              unsigned lds, int tile_rows) {
  const unsigned long long ga = (unsigned long long)(__SIZE_TYPE__)gp;
  u32x4 g0;
  g0.x = 1u;                                   // count=1, user mode
  g0.y = lds;                                  // lds_addr [63:32]
  g0.z = (unsigned)ga;                         // global_addr lo
  g0.w = (unsigned)(ga >> 32) | 0x80000000u;   // global_addr hi | type=2
  i32x8 g1;
  g1[0] = 0x00010000;                          // wg_mask=0, data_size=1 (2B)
  g1[1] = (CC & 0xFFFF) << 16;                 // tensor_dim0 lo16 (=768)
  g1[2] = (tile_rows & 0xFFFF) << 16;          // dim0 hi16=0 | tensor_dim1 lo16
  g1[3] = 32 << 16;                            // dim1 hi16=0 | tile_dim0=32
  g1[4] = tile_rows & 0xFFFF;                  // tile_dim1 | tile_dim2=0
  g1[5] = CC;                                  // tensor_dim0_stride lo32
  g1[6] = 0;                                   // stride0 hi | stride1 lo
  g1[7] = 0;
  const i32x4 z4 = {0, 0, 0, 0};
#if __clang_major__ >= 23
  const i32x8 z8 = {0, 0, 0, 0, 0, 0, 0, 0};
  __builtin_amdgcn_tensor_load_to_lds(g0, g1, z4, z4, z8, 0);
#else
  __builtin_amdgcn_tensor_load_to_lds(g0, g1, z4, z4, 0);
#endif
}
#endif

// ---------------------------------------------------------------------------
// Pass 0: fp32 -> bf16, vectorized 4-wide (n4 = elements/4).
// ---------------------------------------------------------------------------
__global__ __launch_bounds__(256) void cvt_f32_bf16(
    const float* __restrict__ s, unsigned short* __restrict__ d, int n4) {
  const int i = blockIdx.x * 256 + threadIdx.x;
  if (i < n4) {
    const f32x4 v = *(const f32x4*)(s + (size_t)i * 4);
    u32x2 p;
    p.x = (unsigned)f2bf(v.x) | ((unsigned)f2bf(v.y) << 16);
    p.y = (unsigned)f2bf(v.z) | ((unsigned)f2bf(v.w) << 16);
    *(u32x2*)(d + (size_t)i * 4) = p;
  }
}

// ---------------------------------------------------------------------------
// Pass 1: qkv = x @ w_qkv^T  (M=8192, N=2304, K=768), all operands bf16.
// Block 64(M) x 128(N), 8 waves of 32x32, K-step 32, TDM double buffering.
// Scatters into head-major Q (scaled 1/8), K, and transposed V [b,h,d,n].
// ---------------------------------------------------------------------------
__global__ __launch_bounds__(256) void qkv_gemm(
    const unsigned short* __restrict__ xb, const unsigned short* __restrict__ wb,
    unsigned short* __restrict__ qb, unsigned short* __restrict__ kb,
    unsigned short* __restrict__ vb) {
  __shared__ __align__(16) unsigned short As[2][64][32];
  __shared__ __align__(16) unsigned short Bs[2][128][32];

  const int tid  = threadIdx.x;
  const int wave = tid >> 5, lane = tid & 31;
  const int wm = wave >> 2, wn = wave & 3;
  const int r  = lane & 15, hh = lane >> 4;
  const int m0 = blockIdx.y * 64;
  const int n0 = blockIdx.x * 128;

  const f32x8 zero = {0.f,0.f,0.f,0.f,0.f,0.f,0.f,0.f};
  f32x8 acc[2][2];
  acc[0][0] = zero; acc[0][1] = zero; acc[1][0] = zero; acc[1][1] = zero;

  const unsigned short* abase = xb + (size_t)m0 * CC;
  const unsigned short* bbase = wb + (size_t)n0 * CC;

#if USE_TDM
  if (wave == 0) {
    tdm_load_tile(abase, lds_off(&As[0][0][0]), 64);
    tdm_load_tile(bbase, lds_off(&Bs[0][0][0]), 128);
    __builtin_amdgcn_s_wait_tensorcnt(0);
  }
#else
  *(u32x4*)&As[0][tid >> 2][(tid & 3) * 8] =
      *(const u32x4*)(abase + (size_t)(tid >> 2) * CC + (tid & 3) * 8);
  {
    const unsigned short* src = bbase + (size_t)(tid >> 1) * CC + (tid & 1) * 16;
    *(u32x4*)&Bs[0][tid >> 1][(tid & 1) * 16]     = *(const u32x4*)(src);
    *(u32x4*)&Bs[0][tid >> 1][(tid & 1) * 16 + 8] = *(const u32x4*)(src + 8);
  }
#endif
  __syncthreads();

  int cur = 0;
  for (int k0 = 0; k0 < CC; k0 += 32) {
    const int nxt = cur ^ 1;
    if (k0 + 32 < CC) {
#if USE_TDM
      if (wave == 0) {
        tdm_load_tile(abase + k0 + 32, lds_off(&As[nxt][0][0]), 64);
        tdm_load_tile(bbase + k0 + 32, lds_off(&Bs[nxt][0][0]), 128);
      }
#else
      *(u32x4*)&As[nxt][tid >> 2][(tid & 3) * 8] =
          *(const u32x4*)(abase + (size_t)(tid >> 2) * CC + k0 + 32 + (tid & 3) * 8);
      const unsigned short* src =
          bbase + (size_t)(tid >> 1) * CC + k0 + 32 + (tid & 1) * 16;
      *(u32x4*)&Bs[nxt][tid >> 1][(tid & 1) * 16]     = *(const u32x4*)(src);
      *(u32x4*)&Bs[nxt][tid >> 1][(tid & 1) * 16 + 8] = *(const u32x4*)(src + 8);
#endif
    }

    Frag af[2], bfr[2];
#pragma unroll
    for (int s = 0; s < 2; ++s) {
      const unsigned short* ap = &As[cur][wm * 32 + s * 16 + r][0];
      af[s].q[0] = *(const u32x4*)(ap + hh * 8);
      af[s].q[1] = *(const u32x4*)(ap + 16 + hh * 8);
      const unsigned short* bp = &Bs[cur][wn * 32 + s * 16 + r][0];
      bfr[s].q[0] = *(const u32x4*)(bp + hh * 8);
      bfr[s].q[1] = *(const u32x4*)(bp + 16 + hh * 8);
    }
#pragma unroll
    for (int i = 0; i < 2; ++i)
#pragma unroll
      for (int j = 0; j < 2; ++j)
        acc[i][j] = wmma_bf16(af[i], bfr[j], acc[i][j]);

#if USE_TDM
    if (wave == 0) __builtin_amdgcn_s_wait_tensorcnt(0);
#endif
    __syncthreads();
    cur = nxt;
  }

  // Epilogue: routing is uniform per 16-col tile -> branch once, then 8
  // strided b16 stores.  (m-tiles never straddle a batch boundary: 1024%16==0)
#pragma unroll
  for (int si = 0; si < 2; ++si)
#pragma unroll
    for (int sj = 0; sj < 2; ++sj) {
      const int jg    = n0 + wn * 32 + sj * 16 + r;    // 0..2303
      const int which = jg / CC;                       // 0=q 1=k 2=v
      const int c     = jg - which * CC;
      const int h     = c >> 6, d = c & 63;
      const int mg0   = m0 + wm * 32 + si * 16 + hh * 8;
      const int b     = mg0 >> 10, nb = mg0 & 1023;
      unsigned short* dst;
      int   stride;
      float scale;
      if (which == 0) {
        dst = qb + ((size_t)(b * NH + h) * NN + nb) * HD + d;
        stride = HD; scale = 0.125f;                   // 64^-0.5
      } else if (which == 1) {
        dst = kb + ((size_t)(b * NH + h) * NN + nb) * HD + d;
        stride = HD; scale = 1.f;
      } else {
        dst = vb + ((size_t)(b * NH + h) * HD + d) * NN + nb;
        stride = 1;  scale = 1.f;
      }
#pragma unroll
      for (int i = 0; i < 8; ++i)
        dst[(size_t)i * stride] = f2bf(acc[si][sj][i] * scale);
    }
}

// ---------------------------------------------------------------------------
// Pass 2: flash attention per (b,h).  One wave = 16 queries; Q fragments in
// registers; 32-key K/V tiles streamed from global (L2-resident); online
// softmax via wave32 shuffles; P D-layout -> A-layout via 1KB per-wave LDS.
// ---------------------------------------------------------------------------
__global__ __launch_bounds__(128) void attn_fused(
    const unsigned short* __restrict__ qb, const unsigned short* __restrict__ kb,
    const unsigned short* __restrict__ vb, unsigned short* __restrict__ ob) {
  __shared__ __align__(16) unsigned short Plds[4][16][32];

  const int bh   = blockIdx.y;                 // b*12 + h
  const int wave = threadIdx.x >> 5, lane = threadIdx.x & 31;
  const int r = lane & 15, hh = lane >> 4;
  const int q0 = blockIdx.x * 64 + wave * 16;

  const unsigned short* qh = qb + (size_t)bh * NN * HD;
  const unsigned short* kh = kb + (size_t)bh * NN * HD;
  const unsigned short* vh = vb + (size_t)bh * HD * NN;   // V^T: [d][n]

  Frag qf[2];
  {
    const unsigned short* qp = qh + (size_t)(q0 + r) * HD;
    qf[0].q[0] = *(const u32x4*)(qp + hh * 8);
    qf[0].q[1] = *(const u32x4*)(qp + 16 + hh * 8);
    qf[1].q[0] = *(const u32x4*)(qp + 32 + hh * 8);
    qf[1].q[1] = *(const u32x4*)(qp + 48 + hh * 8);
  }

  const f32x8 zero = {0.f,0.f,0.f,0.f,0.f,0.f,0.f,0.f};
  f32x8 O[4];
  O[0] = zero; O[1] = zero; O[2] = zero; O[3] = zero;
  float rmax[8], rsum[8];
#pragma unroll
  for (int i = 0; i < 8; ++i) { rmax[i] = -3.0e38f; rsum[i] = 0.f; }

  for (int kt = 0; kt < NN; kt += 32) {
    Frag kf[2][2];
#pragma unroll
    for (int s = 0; s < 2; ++s) {
      const unsigned short* kp = kh + (size_t)(kt + s * 16 + r) * HD;
      kf[s][0].q[0] = *(const u32x4*)(kp + hh * 8);
      kf[s][0].q[1] = *(const u32x4*)(kp + 16 + hh * 8);
      kf[s][1].q[0] = *(const u32x4*)(kp + 32 + hh * 8);
      kf[s][1].q[1] = *(const u32x4*)(kp + 48 + hh * 8);
    }
    f32x8 S0 = zero, S1 = zero;
    S0 = wmma_bf16(qf[0], kf[0][0], S0);
    S0 = wmma_bf16(qf[1], kf[0][1], S0);
    S1 = wmma_bf16(qf[0], kf[1][0], S1);
    S1 = wmma_bf16(qf[1], kf[1][1], S1);

    float corr[8];
#pragma unroll
    for (int i = 0; i < 8; ++i) {
      float t = fmaxf(S0[i], S1[i]);
      t = fmaxf(t, __shfl_xor(t, 1, 32));
      t = fmaxf(t, __shfl_xor(t, 2, 32));
      t = fmaxf(t, __shfl_xor(t, 4, 32));
      t = fmaxf(t, __shfl_xor(t, 8, 32));
      const float nm = fmaxf(rmax[i], t);
      corr[i] = __expf(rmax[i] - nm);
      rmax[i] = nm;
      const float p0 = __expf(S0[i] - nm);
      const float p1 = __expf(S1[i] - nm);
      float rs = p0 + p1;
      rs += __shfl_xor(rs, 1, 32);
      rs += __shfl_xor(rs, 2, 32);
      rs += __shfl_xor(rs, 4, 32);
      rs += __shfl_xor(rs, 8, 32);
      rsum[i] = rsum[i] * corr[i] + rs;
      Plds[wave][hh * 8 + i][r]      = f2bf(p0);
      Plds[wave][hh * 8 + i][16 + r] = f2bf(p1);
    }

    Frag pf;                                   // same-wave LDS ops are in-order
    pf.q[0] = *(const u32x4*)(&Plds[wave][r][hh * 8]);
    pf.q[1] = *(const u32x4*)(&Plds[wave][r][16 + hh * 8]);

#pragma unroll
    for (int c = 0; c < 4; ++c) {
#pragma unroll
      for (int i = 0; i < 8; ++i) O[c][i] *= corr[i];
      Frag vf;
      const unsigned short* vp = vh + (size_t)(c * 16 + r) * NN + kt;
      vf.q[0] = *(const u32x4*)(vp + hh * 8);
      vf.q[1] = *(const u32x4*)(vp + 16 + hh * 8);
      O[c] = wmma_bf16(pf, vf, O[c]);
    }
  }

  const int b = bh / NH, h = bh - b * NH;
#pragma unroll
  for (int i = 0; i < 8; ++i) {
    const float inv = 1.0f / rsum[i];
    const int row = q0 + hh * 8 + i;
    unsigned short* op = ob + (size_t)(b * NN + row) * CC + h * HD;
#pragma unroll
    for (int c = 0; c < 4; ++c)
      op[c * 16 + r] = f2bf(O[c][i] * inv);
  }
}

// ---------------------------------------------------------------------------
// Pass 3: out = attn @ w_proj^T + b_proj (M=8192, N=768, K=768), fp32 out.
// ---------------------------------------------------------------------------
__global__ __launch_bounds__(256) void proj_gemm(
    const unsigned short* __restrict__ ab, const unsigned short* __restrict__ wb,
    const float* __restrict__ bias, float* __restrict__ out) {
  __shared__ __align__(16) unsigned short As[2][64][32];
  __shared__ __align__(16) unsigned short Bs[2][128][32];

  const int tid  = threadIdx.x;
  const int wave = tid >> 5, lane = tid & 31;
  const int wm = wave >> 2, wn = wave & 3;
  const int r  = lane & 15, hh = lane >> 4;
  const int m0 = blockIdx.y * 64;
  const int n0 = blockIdx.x * 128;

  const f32x8 zero = {0.f,0.f,0.f,0.f,0.f,0.f,0.f,0.f};
  f32x8 acc[2][2];
  acc[0][0] = zero; acc[0][1] = zero; acc[1][0] = zero; acc[1][1] = zero;

  const unsigned short* abase = ab + (size_t)m0 * CC;
  const unsigned short* bbase = wb + (size_t)n0 * CC;

#if USE_TDM
  if (wave == 0) {
    tdm_load_tile(abase, lds_off(&As[0][0][0]), 64);
    tdm_load_tile(bbase, lds_off(&Bs[0][0][0]), 128);
    __builtin_amdgcn_s_wait_tensorcnt(0);
  }
#else
  *(u32x4*)&As[0][tid >> 2][(tid & 3) * 8] =
      *(const u32x4*)(abase + (size_t)(tid >> 2) * CC + (tid & 3) * 8);
  {
    const unsigned short* src = bbase + (size_t)(tid >> 1) * CC + (tid & 1) * 16;
    *(u32x4*)&Bs[0][tid >> 1][(tid & 1) * 16]     = *(const u32x4*)(src);
    *(u32x4*)&Bs[0][tid >> 1][(tid & 1) * 16 + 8] = *(const u32x4*)(src + 8);
  }
#endif
  __syncthreads();

  int cur = 0;
  for (int k0 = 0; k0 < CC; k0 += 32) {
    const int nxt = cur ^ 1;
    if (k0 + 32 < CC) {
#if USE_TDM
      if (wave == 0) {
        tdm_load_tile(abase + k0 + 32, lds_off(&As[nxt][0][0]), 64);
        tdm_load_tile(bbase + k0 + 32, lds_off(&Bs[nxt][0][0]), 128);
      }
#else
      *(u32x4*)&As[nxt][tid >> 2][(tid & 3) * 8] =
          *(const u32x4*)(abase + (size_t)(tid >> 2) * CC + k0 + 32 + (tid & 3) * 8);
      const unsigned short* src =
          bbase + (size_t)(tid >> 1) * CC + k0 + 32 + (tid & 1) * 16;
      *(u32x4*)&Bs[nxt][tid >> 1][(tid & 1) * 16]     = *(const u32x4*)(src);
      *(u32x4*)&Bs[nxt][tid >> 1][(tid & 1) * 16 + 8] = *(const u32x4*)(src + 8);
#endif
    }

    Frag af[2], bfr[2];
#pragma unroll
    for (int s = 0; s < 2; ++s) {
      const unsigned short* ap = &As[cur][wm * 32 + s * 16 + r][0];
      af[s].q[0] = *(const u32x4*)(ap + hh * 8);
      af[s].q[1] = *(const u32x4*)(ap + 16 + hh * 8);
      const unsigned short* bp = &Bs[cur][wn * 32 + s * 16 + r][0];
      bfr[s].q[0] = *(const u32x4*)(bp + hh * 8);
      bfr[s].q[1] = *(const u32x4*)(bp + 16 + hh * 8);
    }
#pragma unroll
    for (int i = 0; i < 2; ++i)
#pragma unroll
      for (int j = 0; j < 2; ++j)
        acc[i][j] = wmma_bf16(af[i], bfr[j], acc[i][j]);

#if USE_TDM
    if (wave == 0) __builtin_amdgcn_s_wait_tensorcnt(0);
#endif
    __syncthreads();
    cur = nxt;
  }

#pragma unroll
  for (int si = 0; si < 2; ++si)
#pragma unroll
    for (int sj = 0; sj < 2; ++sj) {
      const int   jg = n0 + wn * 32 + sj * 16 + r;
      const float bv = bias[jg];
      const int   mg0 = m0 + wm * 32 + si * 16 + hh * 8;
#pragma unroll
      for (int i = 0; i < 8; ++i)
        out[(size_t)(mg0 + i) * CC + jg] = acc[si][sj][i] + bv;
    }
}

// ---------------------------------------------------------------------------
extern "C" void kernel_launch(void* const* d_in, const int* in_sizes, int n_in,
                              void* d_out, int out_size, void* d_ws, size_t ws_size,
                              hipStream_t stream) {
  const float* x      = (const float*)d_in[0];
  const float* w_qkv  = (const float*)d_in[1];
  const float* w_proj = (const float*)d_in[2];
  const float* b_proj = (const float*)d_in[3];
  float* out = (float*)d_out;

  // Workspace (bf16): Q,K,V^T,attn-out + converted x/w_qkv/w_proj (~68 MB).
  unsigned short* qb  = (unsigned short*)d_ws;
  unsigned short* kb  = qb + QKV_ELEMS;
  unsigned short* vb  = kb + QKV_ELEMS;
  unsigned short* ab  = vb + QKV_ELEMS;
  unsigned short* xb  = ab + ATT_ELEMS;
  unsigned short* wqb = xb + (size_t)MM * CC;
  unsigned short* wpb = wqb + (size_t)3 * CC * CC;

  const int x_n4  = MM * CC / 4;        // 1572864
  const int wq_n4 = 3 * CC * CC / 4;    // 442368
  const int wp_n4 = CC * CC / 4;        // 147456
  cvt_f32_bf16<<<(x_n4  + 255) / 256, 256, 0, stream>>>(x,      xb,  x_n4);
  cvt_f32_bf16<<<(wq_n4 + 255) / 256, 256, 0, stream>>>(w_qkv,  wqb, wq_n4);
  cvt_f32_bf16<<<(wp_n4 + 255) / 256, 256, 0, stream>>>(w_proj, wpb, wp_n4);

  qkv_gemm <<<dim3(3 * CC / 128, MM / 64), 256, 0, stream>>>(xb, wqb, qb, kb, vb);
  attn_fused<<<dim3(NN / 64, BB * NH),     128, 0, stream>>>(qb, kb, vb, ab);
  proj_gemm<<<dim3(CC / 128, MM / 64),     256, 0, stream>>>(ab, wpb, b_proj, out);
}
